// CustomModel_79396765433857
// MI455X (gfx1250) — matrix operand order repaired and checked by reading, not compiled, not run on hardware
//
#include <hip/hip_runtime.h>

typedef __attribute__((ext_vector_type(16))) _Float16 v16h;
typedef __attribute__((ext_vector_type(8)))  _Float16 v8h;
typedef __attribute__((ext_vector_type(8)))  float    v8f;

#define B_      4
#define N_      512
#define T_      10
#define F_      516      // N + 4
#define MSG_HID 128
#define MSG_DIM 32
#define UPD_HID 128
#define OUT_DIM 4

#define KP      64       // padded K for GEMM1: 0..39 base, 40..49 extra, 50 = bias col
#define AP      72       // As row stride (f16): 144 B -> 16B aligned, banks decorrelate
#define W1S     72       // W1t row stride: 144 B
#define W2S     136      // W2t row stride: 272 B
#define HSP     136      // Hs row stride: 272 B

union VF  { v16h v; uint4 q[2]; };
union PK8 { v8h v8; _Float16 h[8]; uint4 q; };
union PK4 { _Float16 h[4]; uint2 u; };
union PK2 { _Float16 h[2]; unsigned u; };

// -------------------------------------------------------------------------
// Kernel 1: per (b,i) block computes msg_sum[b,i,:] = sum_j msg(i,j).
// Transposed GEMMs (weights = WMMA-A, data = WMMA-B):
//   h^T = relu(W1^T @ feat^T)   (mb1 folded as K-column 50)
//   msg^T = W2^T @ h^T          (accumulated in WMMA-C across all tiles)
// W1 fragments: nt 0..3 hoisted in registers, nt 4..7 gathered per tile.
// Global x loads pipelined one tile ahead as raw float4s; converted to f16
// at store time so each cvt_pk pair feeds a b64 LDS store directly.
// -------------------------------------------------------------------------
__global__ __launch_bounds__(128) void edge_msg_kernel(
    const float* __restrict__ x,   const float* __restrict__ mW1,
    const float* __restrict__ mb1, const float* __restrict__ mW2,
    const float* __restrict__ mb2, float* __restrict__ msg_out)
{
    __shared__ __attribute__((aligned(16))) _Float16 W1t[MSG_HID][W1S]; // W1^T, K-permuted
    __shared__ __attribute__((aligned(16))) _Float16 W2t[MSG_DIM][W2S]; // W2^T
    __shared__ __attribute__((aligned(16))) _Float16 As [4][16][AP];    // per-wave feat tile
    __shared__ __attribute__((aligned(16))) _Float16 Hs [4][16][HSP];   // per-wave h tile
    __shared__ float msgAcc[MSG_DIM];

    const int tid  = threadIdx.x;
    const int w    = tid >> 5;
    const int lane = tid & 31;
    const int m    = lane & 15;
    const int kh   = lane >> 4;
    const int b    = blockIdx.x / N_;
    const int i    = blockIdx.x - b * N_;

    float4 xv[T_];      // raw prefetched tile, carried across iterations

    auto issue_loads = [&](int tl) {
        const int j = tl * 16 + m;
        const float* xr = x + (size_t)(b * N_ + j) * (T_ * F_);
        if (kh == 0) {
            #pragma unroll
            for (int t = 0; t < T_; ++t) xv[t] = *(const float4*)(xr + t * F_);
        } else {
            #pragma unroll
            for (int t = 0; t < T_; ++t) xv[t].x = xr[t * F_ + 4 + i];
        }
    };

    issue_loads(w);     // first tile's x loads overlap the weight staging below

    // ---- stage W1^T (f16, K-permuted, mb1 folded at kp==50, zero pad) ----
    for (int idx = tid; idx < MSG_HID * KP; idx += 128) {
        const int kp = idx >> 7;              // 0..63
        const int h  = idx & (MSG_HID - 1);   // coalesced over h
        float v;
        if (kp < 40)       v = mW1[((kp >> 2) * 5 + (kp & 3)) * MSG_HID + h];
        else if (kp < 50)  v = mW1[((kp - 40) * 5 + 4) * MSG_HID + h];
        else if (kp == 50) v = mb1[h];
        else               v = 0.0f;
        W1t[h][kp] = (_Float16)v;
    }
    // ---- stage W2^T ----
    for (int idx = tid; idx < MSG_HID * MSG_DIM; idx += 128) {
        const int k = idx >> 5, n = idx & 31;
        W2t[n][k] = (_Float16)mW2[idx];
    }
    if (tid < MSG_DIM) msgAcc[tid] = (float)N_ * mb2[tid];  // fold mb2 * 512
    // constant A-tile columns: bias input (1.0) at 50, zeros at 51..63
    if (kh == 0) {
        As[w][m][50] = (_Float16)1.0f;
        for (int k = 51; k < KP; ++k) As[w][m][k] = (_Float16)0.0f;
    }
    __syncthreads();

    // ---- hoist W1^T A-fragments for nt = 0..3 only (64 VGPRs) ----
    v16h w1f[4][2];
    #pragma unroll
    for (int nt = 0; nt < 4; ++nt) {
        #pragma unroll
        for (int ks = 0; ks < 2; ++ks) {
            const _Float16* p = &W1t[nt * 16 + m][ks * 32 + kh * 8];
            VF f;
            f.q[0] = *(const uint4*)p;          // elements 0..7  (K = base+0..7)
            f.q[1] = *(const uint4*)(p + 16);   // elements 8..15 (K = base+16..23)
            w1f[nt][ks] = f.v;
        }
    }

    v8f c0 = {}, c1 = {};   // msg^T accumulators, live across all tiles

    #pragma unroll 1
    for (int tile = w; tile < N_ / 16; tile += 4) {
        const bool more = (tile + 4) < (N_ / 16);

        // ---- convert + spill the prefetched tile (cvt_pk pairs -> b64/b32) ----
        if (kh == 0) {
            #pragma unroll
            for (int t = 0; t < T_; ++t) {
                PK4 p;
                p.h[0] = (_Float16)xv[t].x; p.h[1] = (_Float16)xv[t].y;
                p.h[2] = (_Float16)xv[t].z; p.h[3] = (_Float16)xv[t].w;
                *(uint2*)&As[w][m][t * 4] = p.u;            // b64 stores
            }
        } else {
            #pragma unroll
            for (int p2 = 0; p2 < 5; ++p2) {
                PK2 e;
                e.h[0] = (_Float16)xv[2 * p2].x;
                e.h[1] = (_Float16)xv[2 * p2 + 1].x;
                *(unsigned*)&As[w][m][40 + 2 * p2] = e.u;   // b32 stores
            }
        }
        __asm__ volatile("s_wait_dscnt 0" ::: "memory");

        // ---- feat^T B-fragments: per lane K = kh*8+0..7 and 16+kh*8+0..7 -> 2 x b128 ----
        VF blo, bhi;
        {
            const _Float16* p = &As[w][m][kh * 8];
            blo.q[0] = *(const uint4*)p;
            blo.q[1] = *(const uint4*)(p + 16);
            bhi.q[0] = *(const uint4*)(p + 32);
            bhi.q[1] = *(const uint4*)(p + 48);
        }

        // ---- issue next tile's global loads; consumed at next loop head ----
        if (more) issue_loads(tile + 4);

        // ---- GEMM1: h^T = relu(W1^T @ feat^T); packed-f16 relu, b128 store ----
        #pragma unroll
        for (int nt = 0; nt < 8; ++nt) {
            v16h a_lo, a_hi;
            if (nt < 4) {                        // compile-time split
                a_lo = w1f[nt][0];
                a_hi = w1f[nt][1];
            } else {                             // gather from LDS per tile
                const _Float16* p = &W1t[nt * 16 + m][kh * 8];
                VF f0, f1;
                f0.q[0] = *(const uint4*)p;
                f0.q[1] = *(const uint4*)(p + 16);
                f1.q[0] = *(const uint4*)(p + 32);
                f1.q[1] = *(const uint4*)(p + 48);
                a_lo = f0.v;
                a_hi = f1.v;
            }
            v8f c = {};
            c = __builtin_amdgcn_wmma_f32_16x16x32_f16(false, a_lo, false, blo.v,
                                                       (short)0, c, false, false);
            c = __builtin_amdgcn_wmma_f32_16x16x32_f16(false, a_hi, false, bhi.v,
                                                       (short)0, c, false, false);
            PK8 pk;
            #pragma unroll
            for (int r = 0; r < 8; ++r) pk.h[r] = (_Float16)c[r];
            const v8h zero = {};
            pk.v8 = __builtin_elementwise_max(pk.v8, zero);   // v_pk_max_num_f16 x4
            // lane holds edge m, hidden nt*16 + kh*8 + 0..7 (contiguous)
            *(uint4*)&Hs[w][m][nt * 16 + kh * 8] = pk.q;
        }
        __asm__ volatile("s_wait_dscnt 0" ::: "memory");

        // ---- GEMM2: msg^T += W2^T @ h^T (4 K-steps, 2 msg-col tiles) ----
        #pragma unroll
        for (int ks = 0; ks < 4; ++ks) {
            VF b2, a0, a1;
            const _Float16* ph = &Hs[w][m][ks * 32 + kh * 8];
            b2.q[0] = *(const uint4*)ph;
            b2.q[1] = *(const uint4*)(ph + 16);
            const _Float16* p0 = &W2t[m][ks * 32 + kh * 8];
            a0.q[0] = *(const uint4*)p0;
            a0.q[1] = *(const uint4*)(p0 + 16);
            const _Float16* p1 = &W2t[16 + m][ks * 32 + kh * 8];
            a1.q[0] = *(const uint4*)p1;
            a1.q[1] = *(const uint4*)(p1 + 16);
            c0 = __builtin_amdgcn_wmma_f32_16x16x32_f16(false, a0.v, false, b2.v,
                                                        (short)0, c0, false, false);
            c1 = __builtin_amdgcn_wmma_f32_16x16x32_f16(false, a1.v, false, b2.v,
                                                        (short)0, c1, false, false);
        }
    }

    // ---- reduce over edges: lane holds (msgcol = kh*8+r [+16], edge = m) ----
    #pragma unroll
    for (int r = 0; r < 8; ++r) {
        atomicAdd(&msgAcc[kh * 8 + r],      c0[r]);
        atomicAdd(&msgAcc[16 + kh * 8 + r], c1[r]);
    }
    __syncthreads();
    if (tid < MSG_DIM)
        msg_out[(size_t)blockIdx.x * MSG_DIM + tid] = msgAcc[tid];
}

// -------------------------------------------------------------------------
// Kernel 2: per-node update MLP (36 -> 128 relu -> 4). 2048 nodes total.
// -------------------------------------------------------------------------
__global__ __launch_bounds__(128) void node_update_kernel(
    const float* __restrict__ x,   const float* __restrict__ msg_sum,
    const float* __restrict__ iW1, const float* __restrict__ ib1,
    const float* __restrict__ iW2, const float* __restrict__ ib2,
    float* __restrict__ out)
{
    __shared__ float W1s[MSG_DIM + 4][UPD_HID];   // 36 x 128
    __shared__ float W2s[UPD_HID][OUT_DIM];
    __shared__ float b1s[UPD_HID];
    __shared__ float b2s[OUT_DIM];

    const int tid = threadIdx.x;
    for (int idx = tid; idx < (MSG_DIM + 4) * UPD_HID; idx += 128)
        W1s[idx >> 7][idx & 127] = iW1[idx];
    for (int idx = tid; idx < UPD_HID * OUT_DIM; idx += 128)
        W2s[idx >> 2][idx & 3] = iW2[idx];
    if (tid < UPD_HID) b1s[tid] = ib1[tid];
    if (tid < OUT_DIM) b2s[tid] = ib2[tid];
    __syncthreads();

    const int node = blockIdx.x * 128 + tid;      // 0 .. B*N-1
    const int b = node / N_, n = node % N_;

    float mi[MSG_DIM + 4];
    #pragma unroll
    for (int k = 0; k < MSG_DIM; ++k) mi[k] = msg_sum[(size_t)node * MSG_DIM + k];
    const float* xf = x + (((size_t)(b * N_ + n)) * T_ + (T_ - 1)) * F_;
    #pragma unroll
    for (int q = 0; q < 4; ++q) mi[MSG_DIM + q] = xf[q];

    float acc[OUT_DIM];
    #pragma unroll
    for (int o = 0; o < OUT_DIM; ++o) acc[o] = b2s[o];
    for (int h = 0; h < UPD_HID; ++h) {
        float s = b1s[h];
        #pragma unroll
        for (int k = 0; k < MSG_DIM + 4; ++k) s += mi[k] * W1s[k][h];
        s = s > 0.0f ? s : 0.0f;
        #pragma unroll
        for (int o = 0; o < OUT_DIM; ++o) acc[o] += s * W2s[h][o];
    }
    #pragma unroll
    for (int o = 0; o < OUT_DIM; ++o)
        out[(size_t)node * OUT_DIM + o] = acc[o];
}

// -------------------------------------------------------------------------
extern "C" void kernel_launch(void* const* d_in, const int* in_sizes, int n_in,
                              void* d_out, int out_size, void* d_ws, size_t ws_size,
                              hipStream_t stream) {
    const float* x   = (const float*)d_in[0];
    const float* mW1 = (const float*)d_in[1];
    const float* mb1 = (const float*)d_in[2];
    const float* mW2 = (const float*)d_in[3];
    const float* mb2 = (const float*)d_in[4];
    const float* iW1 = (const float*)d_in[5];
    const float* ib1 = (const float*)d_in[6];
    const float* iW2 = (const float*)d_in[7];
    const float* ib2 = (const float*)d_in[8];
    float* out = (float*)d_out;
    float* msg = (float*)d_ws;                 // B*N*32 floats = 256 KB scratch

    edge_msg_kernel<<<B_ * N_, 128, 0, stream>>>(x, mW1, mb1, mW2, mb2, msg);
    node_update_kernel<<<(B_ * N_) / 128, 128, 0, stream>>>(x, msg, iW1, ib1,
                                                            iW2, ib2, out);
}
